// SegVFE_6614249636213
// MI455X (gfx1250) — compile-verified, matrix operand order checked by reading, not run on hardware
//
#include <hip/hip_runtime.h>
#include <hip/hip_bf16.h>

// ---------------------------------------------------------------------------
// SegVFE on gfx1250 (MI455X): fused point-MLP with v_wmma_f32_16x16x32_f16,
// BN folded into f16 weights, segment-max via sign-split int atomic max / uint atomic min,
// 256->16 compression kernel. Wave32 / WGP, 318KB LDS per workgroup.
// Round 2: templated layers (full unroll, immediate ds offsets) + B-fragment
// register caching (each wave pins a column tile, sweeps 8 row tiles).
// ---------------------------------------------------------------------------

typedef __attribute__((ext_vector_type(16))) _Float16 v16h;
typedef __attribute__((ext_vector_type(8)))  _Float16 v8h;
typedef __attribute__((ext_vector_type(8)))  float    v8f;

#define EPSB 1e-5f
// voxel_size / offsets are compile-time constants of the reference problem
#define VX   (50.0f / 479.0f)
#define VY   (6.28318530718f / 359.0f)
#define VZ   (6.0f / 31.0f)
#define OFFX (VX * 0.5f + 0.0f)
#define OFFY (VY * 0.5f - 3.14159265359f)
#define OFFZ (VZ * 0.5f - 4.0f)

// --- folded-weight layout in workspace / LDS (halfs), transposed Wt[out][K] --
#define OWT0 0            // [64][32]   (K 9 padded to 32)
#define OWT1 2048         // [128][64]
#define OWT2 10240        // [256][128]
#define OWT3 43008        // [256][256]
#define WT_HALFS 108544   // total halfs
#define BIAS_FLOATS 704   // b0(64) b1(128) b2(256) b3(256), contiguous after wt
#define WBYTES (WT_HALFS * 2 + BIAS_FLOATS * 4)   // 219,904 B, one LDS memcpy

// --- LDS carve (bytes) ------------------------------------------------------
#define L_BIAS  217088
#define L_BUFA  219904    // 128x128 f16 max (also holds X0 128x32)
#define L_BUFB  252672    // 128x256 f16 max
#define L_VOX   318208    // 128 ints
#define L_S0    318720    // 9+9 floats BN0 scale/shift
#define LDS_TOTAL 318800  // <= 320KB WGP LDS

// ---------------------------------------------------------------------------
// sign-split float atomic max -> single global_atomic_max_i32 / min_u32
__device__ __forceinline__ void atomicMaxF(float* addr, float v) {
    if (v >= 0.0f) atomicMax((int*)addr, __float_as_int(v));
    else           atomicMin((unsigned int*)addr, __float_as_uint(v));
}

// Load one 16x32 f16 fragment (A, or B on transposed weights) from LDS.
// lane[3:0] = row (A) / column (B); lane[4] selects K-halves 0..7/16..23 vs
// 8..15/24..31 per the CDNA5 16-bit A-matrix VGPR table. Two ds_load_b128.
template<int STRIDE>
__device__ __forceinline__ v16h load_frag(const _Float16* base, int row,
                                          int k0, int lane) {
    int r   = row + (lane & 15);
    int sel = lane >> 4;
    const _Float16* p = base + r * STRIDE + k0 + sel * 8;
    v8h lo = *(const v8h*)(p);
    v8h hi = *(const v8h*)(p + 16);
    v16h f;
#pragma unroll
    for (int i = 0; i < 8; ++i) { f[i] = lo[i]; f[i + 8] = hi[i]; }
    return f;
}

// One fused Linear(+bias)+ReLU layer: X[128,K] x Wt[OUT,K]^T -> Xout[128,OUT].
// Wave schedule: each job = (row-group, column-tile); the wave caches ALL
// K-chunks of the B fragment in VGPRs and sweeps its row tiles, so weights
// are read from LDS once per 8/RS output tiles. RS=2 splits rows when there
// are fewer column tiles than waves (layer 0).
template<int K, int OUT, int RS, bool RELU>
__device__ __forceinline__ void layer_wmma(const _Float16* Xin,
                                           const _Float16* Wt,
                                           const float* bias, _Float16* Xout,
                                           int lane, int wave) {
    constexpr int CT  = OUT >> 4;     // column tiles
    constexpr int RT  = 8 / RS;       // row tiles per job
    constexpr int KC  = K / 32;       // K chunks
    constexpr int JOBS = CT * RS;     // multiple of 8 for all layers
    const int sel = lane >> 4;
    for (int job = wave; job < JOBS; job += 8) {
        const int ct = job & (CT - 1);
        const int rg = job / CT;
        const int c0 = ct << 4;
        const int col = c0 + (lane & 15);
        const float bval = bias[col];
        v16h bfrag[KC];
#pragma unroll
        for (int k = 0; k < KC; ++k)
            bfrag[k] = load_frag<K>(Wt, c0, k * 32, lane);
#pragma unroll
        for (int rt = 0; rt < RT; ++rt) {
            const int r0 = (rg * RT + rt) << 4;
            v8f acc;
#pragma unroll
            for (int i = 0; i < 8; ++i) acc[i] = bval;
#pragma unroll
            for (int k = 0; k < KC; ++k) {
                v16h a = load_frag<K>(Xin, r0, k * 32, lane);
                acc = __builtin_amdgcn_wmma_f32_16x16x32_f16(
                    false, a, false, bfrag[k], (short)0, acc, false, false);
            }
#pragma unroll
            for (int v = 0; v < 8; ++v) {           // C: row = v + 8*lane[4]
                float x = acc[v];
                if (RELU) x = fmaxf(x, 0.0f);
                Xout[(r0 + v + sel * 8) * OUT + col] = (_Float16)x;
            }
        }
    }
}

// Last layer (plain Linear, K=OUT=256) -> per-voxel max-scatter (segment_max)
__device__ __forceinline__ void layer_final(const _Float16* Xin,
                                            const _Float16* Wt, const float* bias,
                                            const int* voxIds, float* voxbuf,
                                            int base, int N, int lane, int wave) {
    constexpr int K = 256, KC = 8;
    const int sel = lane >> 4;
    for (int job = wave; job < 16; job += 8) {      // 16 column tiles
        const int c0 = job << 4;
        const int col = c0 + (lane & 15);
        const float bval = bias[col];
        v16h bfrag[KC];
#pragma unroll
        for (int k = 0; k < KC; ++k)
            bfrag[k] = load_frag<K>(Wt, c0, k * 32, lane);
#pragma unroll
        for (int rt = 0; rt < 8; ++rt) {
            const int r0 = rt << 4;
            v8f acc;
#pragma unroll
            for (int i = 0; i < 8; ++i) acc[i] = bval;
#pragma unroll
            for (int k = 0; k < KC; ++k) {
                v16h a = load_frag<K>(Xin, r0, k * 32, lane);
                acc = __builtin_amdgcn_wmma_f32_16x16x32_f16(
                    false, a, false, bfrag[k], (short)0, acc, false, false);
            }
#pragma unroll
            for (int v = 0; v < 8; ++v) {
                int row = r0 + v + sel * 8;
                int p = base + row;
                if (p < N)
                    atomicMaxF(&voxbuf[(size_t)voxIds[row] * 256 + col], acc[v]);
            }
        }
    }
}

// ---------------------------------------------------------------------------
struct PrepArgs {
    const float *W0, *b0, *W1, *b1, *W2, *b2, *W3, *b3;
    const float *g1, *be1, *mu1, *va1;
    const float *g2, *be2, *mu2, *va2;
    const float *g3, *be3, *mu3, *va3;
    _Float16* wt; float* bias;
};

// Fold BN into weights, transpose to Wt[out][K] f16, pad layer0 K to 32.
__global__ void seg_prep(PrepArgs p) {
    int t = blockIdx.x * 256 + threadIdx.x;
    if (t < OWT1) {                                   // Wt0 [64][32]
        int out = t >> 5, k = t & 31;
        float s = p.g1[out] * rsqrtf(p.va1[out] + EPSB);
        p.wt[t] = (_Float16)((k < 9) ? p.W0[k * 64 + out] * s : 0.0f);
    } else if (t < OWT2) {                            // Wt1 [128][64]
        int u = t - OWT1; int out = u >> 6, k = u & 63;
        float s = p.g2[out] * rsqrtf(p.va2[out] + EPSB);
        p.wt[t] = (_Float16)(p.W1[k * 128 + out] * s);
    } else if (t < OWT3) {                            // Wt2 [256][128]
        int u = t - OWT2; int out = u >> 7, k = u & 127;
        float s = p.g3[out] * rsqrtf(p.va3[out] + EPSB);
        p.wt[t] = (_Float16)(p.W2[k * 256 + out] * s);
    } else if (t < WT_HALFS) {                        // Wt3 [256][256], no BN
        int u = t - OWT3; int out = u >> 8, k = u & 255;
        p.wt[t] = (_Float16)(p.W3[k * 256 + out]);
    } else if (t < WT_HALFS + BIAS_FLOATS) {          // folded biases
        int u = t - WT_HALFS;
        if (u < 64) {
            float s = p.g1[u] * rsqrtf(p.va1[u] + EPSB);
            p.bias[u] = (p.b0[u] - p.mu1[u]) * s + p.be1[u];
        } else if (u < 192) {
            int c = u - 64; float s = p.g2[c] * rsqrtf(p.va2[c] + EPSB);
            p.bias[u] = (p.b1[c] - p.mu2[c]) * s + p.be2[c];
        } else if (u < 448) {
            int c = u - 192; float s = p.g3[c] * rsqrtf(p.va3[c] + EPSB);
            p.bias[u] = (p.b2[c] - p.mu3[c]) * s + p.be3[c];
        } else {
            p.bias[u] = p.b3[u - 448];
        }
    }
}

__global__ void seg_init(float* voxbuf, int n) {
    int t = blockIdx.x * 256 + threadIdx.x;
    if (t < n) voxbuf[t] = __int_as_float(0xff800000);   // -inf
}

// ---------------------------------------------------------------------------
struct MainArgs {
    const float* feats; const int* p2v; const int* coors;
    const _Float16* wt;                 // wt + contiguous biases in workspace
    const float *g0, *be0, *mu0, *va0;  // pre_norm BN
    float* voxbuf;
    int N; int nTiles;
};

__global__ void __launch_bounds__(256, 1) seg_vfe_main(MainArgs a) {
    extern __shared__ __align__(16) char smem[];
    _Float16* smemH  = (_Float16*)smem;
    const _Float16* WT0 = smemH + OWT0;
    const _Float16* WT1 = smemH + OWT1;
    const _Float16* WT2 = smemH + OWT2;
    const _Float16* WT3 = smemH + OWT3;
    const float* BIAS = (const float*)(smem + L_BIAS);
    _Float16* bufA = (_Float16*)(smem + L_BUFA);
    _Float16* bufB = (_Float16*)(smem + L_BUFB);
    int*   voxIds = (int*)(smem + L_VOX);
    float* s0 = (float*)(smem + L_S0);
    float* t0 = s0 + 9;

    const int tid = threadIdx.x, lane = tid & 31, wave = tid >> 5;

    // Stage all folded weights + biases into LDS once (persistent block).
    {
        const uint4* src = (const uint4*)a.wt;
        uint4* dst = (uint4*)smem;
        for (int i = tid; i < (WBYTES >> 4); i += 256) dst[i] = src[i];
    }
    if (tid < 9) {  // pre_norm BN scale/shift
        float s = a.g0[tid] * rsqrtf(a.va0[tid] + EPSB);
        s0[tid] = s;
        t0[tid] = a.be0[tid] - a.mu0[tid] * s;
    }
    __syncthreads();

    for (int tile = blockIdx.x; tile < a.nTiles; tile += gridDim.x) {
        const int base = tile << 7;                  // 128 points per tile
        // ---- stage X0[128][32] f16 into bufA: gather + center + pre_norm ----
        for (int r = tid; r < 128; r += 256) {
            int p = base + r;
            float x[9]; int vox = 0;
            if (p < a.N) {
                vox = a.p2v[p];
                const int* c4 = a.coors + vox * 4;   // (batch, x, y, z)
                const float* f = a.feats + (size_t)p * 6;
                float f0 = f[0], f1 = f[1], f2 = f[2];
                x[0] = f0 - ((float)c4[1] * VX + OFFX);
                x[1] = f1 - ((float)c4[2] * VY + OFFY);
                x[2] = f2 - ((float)c4[3] * VZ + OFFZ);
                x[3] = f0; x[4] = f1; x[5] = f2;
                x[6] = f[3]; x[7] = f[4]; x[8] = f[5];
            }
            voxIds[r] = vox;
            _Float16* row = bufA + r * 32;
            if (p < a.N) {
#pragma unroll
                for (int c = 0; c < 9; ++c)
                    row[c] = (_Float16)(x[c] * s0[c] + t0[c]);
            } else {
#pragma unroll
                for (int c = 0; c < 9; ++c) row[c] = (_Float16)0.0f;
            }
#pragma unroll
            for (int c = 9; c < 32; ++c) row[c] = (_Float16)0.0f;   // K pad
        }
        __syncthreads();
        layer_wmma< 32,  64, 2, true>(bufA, WT0, BIAS + 0,   bufB, lane, wave);
        __syncthreads();
        layer_wmma< 64, 128, 1, true>(bufB, WT1, BIAS + 64,  bufA, lane, wave);
        __syncthreads();
        layer_wmma<128, 256, 1, true>(bufA, WT2, BIAS + 192, bufB, lane, wave);
        __syncthreads();
        layer_final(bufB, WT3, BIAS + 448, voxIds, a.voxbuf,
                    base, a.N, lane, wave);
        __syncthreads();                             // before bufA/voxIds reuse
    }
}

// ---------------------------------------------------------------------------
// 256->16 compression: 16 voxels per block, voxel rows staged in LDS,
// -inf (empty voxel) -> 0, dot + bias + ReLU.
__global__ void __launch_bounds__(256) seg_compress(const float* voxbuf,
                                                    const float* Wc,
                                                    const float* bc,
                                                    float* out, int M) {
    __shared__ float rows[16][256];
    __shared__ float wc[256 * 16];
    __shared__ float bcs[16];
    const int tid = threadIdx.x;
    const int m0 = blockIdx.x * 16;
    for (int i = tid; i < 4096; i += 256) wc[i] = Wc[i];
    if (tid < 16) bcs[tid] = bc[tid];
    for (int i = tid; i < 4096; i += 256) {
        int r = i >> 8, k = i & 255;
        int m = m0 + r;
        float v = (m < M) ? voxbuf[(size_t)m * 256 + k] : 0.0f;
        unsigned u = __float_as_uint(v);
        if ((u & 0x7f800000u) == 0x7f800000u) v = 0.0f;   // inf/nan -> 0
        rows[r][k] = v;
    }
    __syncthreads();
    int r = tid >> 4, c = tid & 15;
    int m = m0 + r;
    if (m < M) {
        float s = bcs[c];
#pragma unroll 8
        for (int k = 0; k < 256; ++k) s = fmaf(rows[r][k], wc[k * 16 + c], s);
        out[(size_t)m * 16 + c] = fmaxf(s, 0.0f);
    }
}

// ---------------------------------------------------------------------------
extern "C" void kernel_launch(void* const* d_in, const int* in_sizes, int n_in,
                              void* d_out, int out_size, void* d_ws, size_t ws_size,
                              hipStream_t stream) {
    const float* features = (const float*)d_in[0];
    const int*   p2v      = (const int*)d_in[1];
    const int*   coors    = (const int*)d_in[2];
    const int N = in_sizes[0] / 6;
    const int M = in_sizes[2] / 4;

    // workspace: [voxbuf M*256 f32][wt f16][bias f32] (wt/bias contiguous)
    char* ws = (char*)d_ws;
    size_t voxBytes = (size_t)M * 256 * sizeof(float);
    size_t wOff = (voxBytes + 255) & ~(size_t)255;
    float*    voxbuf  = (float*)ws;
    _Float16* wt      = (_Float16*)(ws + wOff);
    float*    biasAll = (float*)(ws + wOff + (size_t)WT_HALFS * 2);

    PrepArgs P;
    P.W0 = (const float*)d_in[3];  P.b0 = (const float*)d_in[4];
    P.W1 = (const float*)d_in[5];  P.b1 = (const float*)d_in[6];
    P.W2 = (const float*)d_in[7];  P.b2 = (const float*)d_in[8];
    P.W3 = (const float*)d_in[9];  P.b3 = (const float*)d_in[10];
    P.g1 = (const float*)d_in[15]; P.be1 = (const float*)d_in[16];
    P.mu1 = (const float*)d_in[17]; P.va1 = (const float*)d_in[18];
    P.g2 = (const float*)d_in[19]; P.be2 = (const float*)d_in[20];
    P.mu2 = (const float*)d_in[21]; P.va2 = (const float*)d_in[22];
    P.g3 = (const float*)d_in[23]; P.be3 = (const float*)d_in[24];
    P.mu3 = (const float*)d_in[25]; P.va3 = (const float*)d_in[26];
    P.wt = wt; P.bias = biasAll;

    int prepTot = WT_HALFS + BIAS_FLOATS;
    seg_prep<<<(prepTot + 255) / 256, 256, 0, stream>>>(P);

    int voxElems = M * 256;
    seg_init<<<(voxElems + 255) / 256, 256, 0, stream>>>(voxbuf, voxElems);

    MainArgs A;
    A.feats = features; A.p2v = p2v; A.coors = coors; A.wt = wt;
    A.g0 = (const float*)d_in[11]; A.be0 = (const float*)d_in[12];
    A.mu0 = (const float*)d_in[13]; A.va0 = (const float*)d_in[14];
    A.voxbuf = voxbuf; A.N = N; A.nTiles = (N + 127) / 128;

    (void)hipFuncSetAttribute((const void*)seg_vfe_main,
                              hipFuncAttributeMaxDynamicSharedMemorySize,
                              LDS_TOTAL);
    int nBlocks = A.nTiles < 512 ? A.nTiles : 512;
    seg_vfe_main<<<nBlocks, 256, LDS_TOTAL, stream>>>(A);

    const float* Wc = (const float*)d_in[27];
    const float* bc = (const float*)d_in[28];
    float* out = (float*)d_out;
    seg_compress<<<(M + 15) / 16, 256, 0, stream>>>(voxbuf, Wc, bc, out, M);

    // second tuple output: voxel_coors (int32 bits) appended after feats
    hipMemcpyAsync(out + (size_t)M * 16, d_in[2], (size_t)M * 4 * sizeof(int),
                   hipMemcpyDeviceToDevice, stream);
}
